// GPT_446676599274
// MI455X (gfx1250) — matrix operand order repaired and checked by reading, not compiled, not run on hardware
//
#include <hip/hip_runtime.h>
#include <hip/hip_bf16.h>

// ---------------- model constants ----------------
#define BATCH   2
#define SEQ     2048
#define DMODEL  1024
#define NHEADS  16
#define HDIM    64
#define HIDDEN  4096
#define MROWS   (BATCH * SEQ)      // 4096
#define LN_EPS  1e-5f

typedef __attribute__((ext_vector_type(16))) __bf16 v16bf;
typedef __attribute__((ext_vector_type(8)))  __bf16 v8bf;
typedef __attribute__((ext_vector_type(8)))  float  v8f;

// ------------------------------------------------------------------
// Fragment loader: 16x32 bf16 tile (row-major, stride in elems)
// CDNA5 16-bit A/B VGPR layout:
//   lanes 0-15 : row = lane,   K = {0..7} in v[0..3], {16..23} in v[4..7]
//   lanes 16-31: row = lane-16, K offsets +8
// ------------------------------------------------------------------
__device__ __forceinline__ v16bf load_frag(const __bf16* base, int stride) {
  const int lane = threadIdx.x & 31;
  const __bf16* p = base + (size_t)(lane & 15) * stride + ((lane >> 4) << 3);
  v8bf lo = *(const v8bf*)p;
  v8bf hi = *(const v8bf*)(p + 16);
  v16bf r;
#pragma unroll
  for (int i = 0; i < 8; ++i) { r[i] = lo[i]; r[i + 8] = hi[i]; }
  return r;
}

__device__ __forceinline__ v8f wmma_bf16(v16bf a, v16bf b, v8f c) {
  return __builtin_amdgcn_wmma_f32_16x16x32_bf16(
      /*neg_a=*/false, a, /*neg_b=*/false, b,
      /*c_mod=*/(short)0, c, /*reuse_a=*/false, /*reuse_b=*/false);
}

// async global->LDS copy (16B per lane), ASYNCcnt-tracked
__device__ __forceinline__ void async_copy_b128(uint32_t lds_off, const void* gptr) {
  asm volatile("global_load_async_to_lds_b128 %0, %1, off"
               :: "v"(lds_off), "v"((uint64_t)(uintptr_t)gptr) : "memory");
}

// ------------------------------------------------------------------
// Weight convert + transpose: f32 W[K][N] -> bf16 Wt[N][K]
// ------------------------------------------------------------------
__global__ void cvt_t_kernel(const float* __restrict__ in, __bf16* __restrict__ out,
                             int K, int N) {
  size_t idx = (size_t)blockIdx.x * blockDim.x + threadIdx.x;
  if (idx >= (size_t)K * N) return;
  int k = (int)(idx / N), n = (int)(idx % N);
  out[(size_t)n * K + k] = (__bf16)in[idx];
}

// ------------------------------------------------------------------
// LayerNorm: f32 row -> bf16 row
// ------------------------------------------------------------------
__global__ __launch_bounds__(256)
void ln_kernel(const float* __restrict__ x, const float* __restrict__ g,
               const float* __restrict__ b, __bf16* __restrict__ out, int D) {
  const int row = blockIdx.x;
  const float* xr = x + (size_t)row * D;
  float s = 0.f, s2 = 0.f;
  for (int c = threadIdx.x; c < D; c += 256) { float v = xr[c]; s += v; s2 += v * v; }
  __shared__ float rs[256], rs2[256];
  rs[threadIdx.x] = s; rs2[threadIdx.x] = s2;
  __syncthreads();
  for (int o = 128; o > 0; o >>= 1) {
    if (threadIdx.x < o) { rs[threadIdx.x] += rs[threadIdx.x + o]; rs2[threadIdx.x] += rs2[threadIdx.x + o]; }
    __syncthreads();
  }
  const float mu = rs[0] / D;
  const float var = rs2[0] / D - mu * mu;
  const float rv = rsqrtf(var + LN_EPS);
  for (int c = threadIdx.x; c < D; c += 256)
    out[(size_t)row * D + c] = (__bf16)((xr[c] - mu) * rv * g[c] + b[c]);
}

// ------------------------------------------------------------------
// WMMA GEMM: C[M,N] = A[M,K](bf16) x Bt[N,K](bf16)^T  + epilogue
// block tile 128x64, 8 waves, each wave a 32x32 tile (2x2 wmma frags)
// Double-buffered LDS fed by async global->LDS copies (ASYNCcnt),
// last iteration peeled so the steady-state loop is branch-free.
// ------------------------------------------------------------------
#define GA_STRIDE 40   // padded LDS stride (bf16 elems); rows stay 16B aligned
__global__ __launch_bounds__(256)
void gemm_kernel(const __bf16* __restrict__ A, const __bf16* __restrict__ Bt,
                 float* outf, __bf16* outbf,
                 const float* __restrict__ bias, const float* __restrict__ residual,
                 int M, int N, int K, int relu) {
  __shared__ alignas(16) __bf16 As[2][128 * GA_STRIDE];
  __shared__ alignas(16) __bf16 Bs[2][64 * GA_STRIDE];
  const int tid  = threadIdx.x;
  const int lane = tid & 31;
  const int w    = tid >> 5;
  const int wm   = (w & 3) << 5;   // 0,32,64,96
  const int wn   = (w >> 2) << 5;  // 0,32
  const int bm   = blockIdx.x * 128;
  const int bn   = blockIdx.y * 64;

  // per-thread staging slots: A tile = 512 x 16B chunks (2/thread), B = 256 (1/thread)
  const int rA = tid >> 2;              // 0..63 (second A chunk uses rA+64)
  const int c8 = (tid & 3) << 3;        // 0,8,16,24 (bf16 elems)
  const uint32_t asBase = (uint32_t)(uintptr_t)&As[0][0];
  const uint32_t bsBase = (uint32_t)(uintptr_t)&Bs[0][0];
  const uint32_t aBufB  = 128 * GA_STRIDE * 2;   // bytes per A buffer
  const uint32_t bBufB  = 64 * GA_STRIDE * 2;

  auto issue_stage = [&](int buf, int k0) {
    const uint32_t la0 = asBase + (uint32_t)buf * aBufB + (uint32_t)((rA        * GA_STRIDE + c8) * 2);
    const uint32_t la1 = asBase + (uint32_t)buf * aBufB + (uint32_t)(((rA + 64) * GA_STRIDE + c8) * 2);
    const uint32_t lb  = bsBase + (uint32_t)buf * bBufB + (uint32_t)((rA        * GA_STRIDE + c8) * 2);
    async_copy_b128(la0, A  + (size_t)(bm + rA)      * K + k0 + c8);
    async_copy_b128(la1, A  + (size_t)(bm + rA + 64) * K + k0 + c8);
    async_copy_b128(lb,  Bt + (size_t)(bn + rA)      * K + k0 + c8);
  };

  v8f acc[2][2] = {};

  auto compute_tile = [&](int cur) {
    v16bf a0 = load_frag(&As[cur][(wm +  0) * GA_STRIDE], GA_STRIDE);
    v16bf a1 = load_frag(&As[cur][(wm + 16) * GA_STRIDE], GA_STRIDE);
    v16bf b0 = load_frag(&Bs[cur][(wn +  0) * GA_STRIDE], GA_STRIDE);
    v16bf b1 = load_frag(&Bs[cur][(wn + 16) * GA_STRIDE], GA_STRIDE);
    acc[0][0] = wmma_bf16(a0, b0, acc[0][0]);
    acc[0][1] = wmma_bf16(a0, b1, acc[0][1]);
    acc[1][0] = wmma_bf16(a1, b0, acc[1][0]);
    acc[1][1] = wmma_bf16(a1, b1, acc[1][1]);
  };

  issue_stage(0, 0);
  int k0 = 0;
  // steady state: unconditionally issue next stage, wait for current, compute
  for (; k0 + 32 < K; k0 += 32) {
    const int cur = (k0 >> 5) & 1;
    issue_stage(cur ^ 1, k0 + 32);
    asm volatile("s_wait_asynccnt 0x3" ::: "memory");  // stage n landed in LDS
    __syncthreads();
    compute_tile(cur);
    __syncthreads();   // everyone done reading `cur` before it is refilled
  }
  // peeled last tile
  {
    const int cur = (k0 >> 5) & 1;
    asm volatile("s_wait_asynccnt 0x0" ::: "memory");
    __syncthreads();
    compute_tile(cur);
  }

  // epilogue (C layout: VGPR r -> row rowoff+r, col = lane&15)
  const int col = lane & 15;
  const int rowoff = (lane >> 4) << 3;
#pragma unroll
  for (int mi = 0; mi < 2; ++mi)
#pragma unroll
    for (int ni = 0; ni < 2; ++ni)
#pragma unroll
      for (int r = 0; r < 8; ++r) {
        const int rg = bm + wm + mi * 16 + rowoff + r;
        const int cg = bn + wn + ni * 16 + col;
        float v = acc[mi][ni][r];
        if (bias)     v += bias[cg];
        if (relu)     v = fmaxf(v, 0.f);
        if (residual) v += residual[(size_t)rg * N + cg];
        if (outf)  outf[(size_t)rg * N + cg] = v;
        if (outbf) outbf[(size_t)rg * N + cg] = (__bf16)v;
      }
}

// ------------------------------------------------------------------
// Permutes into attention-friendly layouts (bf16 -> bf16)
//   in  : [B*S, H*64]  (row b*S+s, col h*64+d)
// ------------------------------------------------------------------
__global__ void pack_qk_kernel(const __bf16* __restrict__ in, __bf16* __restrict__ out) {
  size_t idx = (size_t)blockIdx.x * blockDim.x + threadIdx.x;
  if (idx >= (size_t)MROWS * DMODEL) return;
  int d = (int)(idx & 63);
  size_t t = idx >> 6;
  int h = (int)(t % NHEADS); t /= NHEADS;
  int s = (int)(t % SEQ);
  int b = (int)(t / SEQ);
  out[(((size_t)b * NHEADS + h) * SEQ + s) * HDIM + d] = in[idx];
}
__global__ void pack_vt_kernel(const __bf16* __restrict__ in, __bf16* __restrict__ out) {
  size_t idx = (size_t)blockIdx.x * blockDim.x + threadIdx.x;
  if (idx >= (size_t)MROWS * DMODEL) return;
  int d = (int)(idx & 63);
  size_t t = idx >> 6;
  int h = (int)(t % NHEADS); t /= NHEADS;
  int s = (int)(t % SEQ);
  int b = (int)(t / SEQ);
  out[(((size_t)b * NHEADS + h) * HDIM + d) * SEQ + s] = in[idx];
}

// ------------------------------------------------------------------
// Flash attention: one wave (32 threads) per 16 query rows per head.
//   Qb,Kb : [B,H,S,64] bf16     Vt : [B,H,64,S] bf16
//   out   : [B,S,H*64] bf16  (ready for the Wo GEMM)
// Online softmax over K tiles of 32 (two 16-wide score frags / iter).
// ------------------------------------------------------------------
__global__ __launch_bounds__(32)
void attn_kernel(const __bf16* __restrict__ Qb, const __bf16* __restrict__ Kb,
                 const __bf16* __restrict__ Vt, __bf16* __restrict__ out) {
  const int qt   = blockIdx.x;          // query tile (16 rows)
  const int bh   = blockIdx.y;          // b*H + h
  const int lane = threadIdx.x & 31;
  const int col  = lane & 15;
  const int rowoff = (lane >> 4) << 3;
  const int q0 = qt * 16;
  const float scale = 0.125f;           // 1/sqrt(64)

  __shared__ alignas(16) __bf16 P[16 * GA_STRIDE];

  const __bf16* qbase = Qb + ((size_t)bh * SEQ + q0) * HDIM;
  const v16bf qa0 = load_frag(qbase,      HDIM);   // K = 0..31 of head dim
  const v16bf qa1 = load_frag(qbase + 32, HDIM);   // K = 32..63

  float mrow[8], lrow[8], alph[8];
#pragma unroll
  for (int r = 0; r < 8; ++r) { mrow[r] = -3.0e38f; lrow[r] = 0.f; }
  v8f o[4] = {};

  for (int j = 0; j <= qt; j += 2) {
    // ---- scores for k tiles j and j+1 ----
    const __bf16* kb0 = Kb + ((size_t)bh * SEQ + j * 16) * HDIM;
    if (j + 2 <= qt)  // hint next K tile pair
      __builtin_prefetch(kb0 + 32 * HDIM, 0, 1);
    v8f c0 = {}, c1 = {};
    c0 = wmma_bf16(qa0, load_frag(kb0,      HDIM), c0);
    c0 = wmma_bf16(qa1, load_frag(kb0 + 32, HDIM), c0);
    const bool has1 = (j + 1) <= qt;
    if (has1) {
      const __bf16* kb1 = kb0 + 16 * HDIM;
      c1 = wmma_bf16(qa0, load_frag(kb1,      HDIM), c1);
      c1 = wmma_bf16(qa1, load_frag(kb1 + 32, HDIM), c1);
    }
    // ---- causal mask + online softmax (rows live across 16-lane halves) ----
#pragma unroll
    for (int r = 0; r < 8; ++r) {
      const int qrow = q0 + rowoff + r;
      float s0 = c0[r] * scale;
      if (j * 16 + col > qrow) s0 = -3.0e38f;
      float s1 = -3.0e38f;
      if (has1) { s1 = c1[r] * scale; if ((j + 1) * 16 + col > qrow) s1 = -3.0e38f; }
      float mt = fmaxf(s0, s1);
#pragma unroll
      for (int m = 1; m < 16; m <<= 1) mt = fmaxf(mt, __shfl_xor(mt, m, 32));
      const float mn = fmaxf(mrow[r], mt);
      const float p0 = __expf(s0 - mn);
      const float p1 = __expf(s1 - mn);
      float ps = p0 + p1;
#pragma unroll
      for (int m = 1; m < 16; m <<= 1) ps += __shfl_xor(ps, m, 32);
      alph[r] = __expf(mrow[r] - mn);
      lrow[r] = lrow[r] * alph[r] + ps;
      mrow[r] = mn;
      P[(rowoff + r) * GA_STRIDE + col]      = (__bf16)p0;
      P[(rowoff + r) * GA_STRIDE + 16 + col] = (__bf16)p1;
    }
    __syncthreads();                       // single-wave: orders DS ops
    const v16bf pa = load_frag(P, GA_STRIDE);
#pragma unroll
    for (int ni = 0; ni < 4; ++ni) {
#pragma unroll
      for (int r = 0; r < 8; ++r) o[ni][r] *= alph[r];
      const __bf16* vb = Vt + ((size_t)bh * HDIM + ni * 16) * SEQ + (size_t)j * 16;
      o[ni] = wmma_bf16(pa, load_frag(vb, SEQ), o[ni]);
    }
    __syncthreads();
  }

  // ---- write normalized output: [B,S,H*64] ----
  const int b = bh / NHEADS, h = bh % NHEADS;
#pragma unroll
  for (int ni = 0; ni < 4; ++ni)
#pragma unroll
    for (int r = 0; r < 8; ++r) {
      const int srow = q0 + rowoff + r;
      out[((size_t)b * SEQ + srow) * DMODEL + h * HDIM + ni * 16 + col] =
          (__bf16)(o[ni][r] / lrow[r]);
    }
}

// ------------------------------------------------------------------
// Host-side orchestration
// ------------------------------------------------------------------
extern "C" void kernel_launch(void* const* d_in, const int* in_sizes, int n_in,
                              void* d_out, int out_size, void* d_ws, size_t ws_size,
                              hipStream_t stream) {
  const float* x   = (const float*)d_in[0];
  const float* wq  = (const float*)d_in[1];
  const float* wk  = (const float*)d_in[2];
  const float* wv  = (const float*)d_in[3];
  const float* wo  = (const float*)d_in[4];
  const float* w1  = (const float*)d_in[5];
  const float* b1  = (const float*)d_in[6];
  const float* w2  = (const float*)d_in[7];
  const float* b2  = (const float*)d_in[8];
  const float* g1  = (const float*)d_in[9];
  const float* be1 = (const float*)d_in[10];
  const float* g2  = (const float*)d_in[11];
  const float* be2 = (const float*)d_in[12];
  float* outp = (float*)d_out;

  // workspace arena
  size_t off = 0;
  char* base = (char*)d_ws;
  auto alloc = [&](size_t bytes) -> void* {
    void* p = base + off;
    off += (bytes + 255) & ~(size_t)255;
    return p;
  };
  __bf16* wqT = (__bf16*)alloc((size_t)DMODEL * DMODEL * 2);
  __bf16* wkT = (__bf16*)alloc((size_t)DMODEL * DMODEL * 2);
  __bf16* wvT = (__bf16*)alloc((size_t)DMODEL * DMODEL * 2);
  __bf16* woT = (__bf16*)alloc((size_t)DMODEL * DMODEL * 2);
  __bf16* w1T = (__bf16*)alloc((size_t)DMODEL * HIDDEN * 2);
  __bf16* w2T = (__bf16*)alloc((size_t)HIDDEN * DMODEL * 2);
  __bf16* A1  = (__bf16*)alloc((size_t)MROWS * DMODEL * 2);
  __bf16* Qf  = (__bf16*)alloc((size_t)MROWS * DMODEL * 2);
  __bf16* Kf  = (__bf16*)alloc((size_t)MROWS * DMODEL * 2);
  __bf16* Vf  = (__bf16*)alloc((size_t)MROWS * DMODEL * 2);
  __bf16* Qb  = (__bf16*)alloc((size_t)MROWS * DMODEL * 2);
  __bf16* Kb  = (__bf16*)alloc((size_t)MROWS * DMODEL * 2);
  __bf16* Vtp = (__bf16*)alloc((size_t)MROWS * DMODEL * 2);
  __bf16* Ab  = (__bf16*)alloc((size_t)MROWS * DMODEL * 2);   // attention out (bf16)
  float*  y1  = (float*) alloc((size_t)MROWS * DMODEL * 4);   // x + attn proj
  __bf16* A2  = (__bf16*)alloc((size_t)MROWS * DMODEL * 2);   // LN2 out
  __bf16* Hb  = (__bf16*)alloc((size_t)MROWS * HIDDEN * 2);   // relu(ffn1) bf16
  (void)ws_size; (void)in_sizes; (void)n_in; (void)out_size;

  const int gDD = (DMODEL * DMODEL) / 256;       // 4096
  const int gDH = (DMODEL * HIDDEN) / 256;       // 16384
  const int gMD = (MROWS * DMODEL) / 256;        // 16384

  // 1) weight transposes -> bf16
  cvt_t_kernel<<<gDD, 256, 0, stream>>>(wq, wqT, DMODEL, DMODEL);
  cvt_t_kernel<<<gDD, 256, 0, stream>>>(wk, wkT, DMODEL, DMODEL);
  cvt_t_kernel<<<gDD, 256, 0, stream>>>(wv, wvT, DMODEL, DMODEL);
  cvt_t_kernel<<<gDD, 256, 0, stream>>>(wo, woT, DMODEL, DMODEL);
  cvt_t_kernel<<<gDH, 256, 0, stream>>>(w1, w1T, DMODEL, HIDDEN);
  cvt_t_kernel<<<gDH, 256, 0, stream>>>(w2, w2T, HIDDEN, DMODEL);

  // 2) LN1 -> bf16
  ln_kernel<<<MROWS, 256, 0, stream>>>(x, g1, be1, A1, DMODEL);

  // 3) Q/K/V projections (bf16 out)
  dim3 gP(MROWS / 128, DMODEL / 64);
  gemm_kernel<<<gP, 256, 0, stream>>>(A1, wqT, nullptr, Qf, nullptr, nullptr,
                                      MROWS, DMODEL, DMODEL, 0);
  gemm_kernel<<<gP, 256, 0, stream>>>(A1, wkT, nullptr, Kf, nullptr, nullptr,
                                      MROWS, DMODEL, DMODEL, 0);
  gemm_kernel<<<gP, 256, 0, stream>>>(A1, wvT, nullptr, Vf, nullptr, nullptr,
                                      MROWS, DMODEL, DMODEL, 0);

  // 4) head-major repacks
  pack_qk_kernel<<<gMD, 256, 0, stream>>>(Qf, Qb);
  pack_qk_kernel<<<gMD, 256, 0, stream>>>(Kf, Kb);
  pack_vt_kernel<<<gMD, 256, 0, stream>>>(Vf, Vtp);

  // 5) flash attention
  dim3 gA(SEQ / 16, BATCH * NHEADS);
  attn_kernel<<<gA, 32, 0, stream>>>(Qb, Kb, Vtp, Ab);

  // 6) output projection + residual: y1 = x + Ab @ Wo
  gemm_kernel<<<gP, 256, 0, stream>>>(Ab, woT, y1, nullptr, nullptr, x,
                                      MROWS, DMODEL, DMODEL, 0);

  // 7) LN2 -> bf16
  ln_kernel<<<MROWS, 256, 0, stream>>>(y1, g2, be2, A2, DMODEL);

  // 8) FFN1: relu(A2 @ W1 + b1) -> bf16
  dim3 gF1(MROWS / 128, HIDDEN / 64);
  gemm_kernel<<<gF1, 256, 0, stream>>>(A2, w1T, nullptr, Hb, b1, nullptr,
                                       MROWS, HIDDEN, DMODEL, 1);

  // 9) FFN2 + bias + residual -> f32 final output
  gemm_kernel<<<gP, 256, 0, stream>>>(Hb, w2T, outp, nullptr, b2, y1,
                                      MROWS, DMODEL, HIDDEN, 0);
}
